// Bottleneck_DLKA_75136157877133
// MI455X (gfx1250) — matrix operand order repaired and checked by reading, not compiled, not run on hardware
//
#include <hip/hip_runtime.h>
#include <hip/hip_bf16.h>

// MI455X / gfx1250, wave32.
// GEMM-shaped work -> v_wmma_f32_16x16x32_bf16 (bf16 in, f32 accumulate).
// Full-M blocking: one workgroup covers all cout blocks (MB accumulators per
// wave) so the im2col B stream is read exactly once. Activations kept as bf16
// mirrors so B bytes are halved. Weights pre-packed to the wave32 A-fragment
// layout and streamed tap-by-tap into LDS with wide copies.

typedef __attribute__((ext_vector_type(16))) __bf16 v16bf;
typedef __attribute__((ext_vector_type(8)))  float  v8f;

#define HH 80
#define WW 80
#define HW 6400
#define NB 4

// ---------------------------------------------------------------------------
// Pack f32 tensor -> bf16 mirror.
// ---------------------------------------------------------------------------
__global__ void pack_bf16_kernel(const float* __restrict__ src,
                                 __bf16* __restrict__ dst, int n)
{
  const int i = blockIdx.x * 256 + threadIdx.x;
  if (i < n) dst[i] = (__bf16)src[i];
}

// ---------------------------------------------------------------------------
// Pack conv weights [cout][cin][k*k] (f32) into bf16 A-fragments, layout
// wpk[tap][cb][mb][lane][e]  (frag = (tap*ncb+cb)*MB + mb, 32 lanes x 16 bf16)
// per ISA 7.12.2 (A 16x32: lanes 0-15 K 0..7/16..23, lanes 16-31 K 8..15/24..31).
// ---------------------------------------------------------------------------
__global__ void pack_w_kernel(const float* __restrict__ w,
                              __bf16* __restrict__ wpk,
                              int cin, int ntap, int cout_real, int MB, int ncb)
{
  const int s = blockIdx.x * 256 + threadIdx.x;
  const int total = ntap * ncb * MB * 32;
  if (s >= total) return;
  const int lane = s & 31;
  const int f    = s >> 5;
  const int mb   = f % MB;
  const int rest = f / MB;
  const int cb   = rest % ncb;
  const int tap  = rest / ncb;
  const int m    = lane & 15;
  const int cout = mb * 16 + m;
  __bf16* dst = wpk + (long)f * 512 + lane * 16;
  if (cout < cout_real) {
    const float* wp = w + (long)cout * cin * ntap + tap;
    #pragma unroll
    for (int e = 0; e < 16; ++e) {
      const int v  = e >> 1;
      const int kl = ((v >= 4) ? 16 : 0) + ((lane >= 16) ? 8 : 0)
                   + ((v & 3) << 1) + (e & 1);
      dst[e] = (__bf16)wp[(long)(cb * 32 + kl) * ntap];
    }
  } else {
    #pragma unroll
    for (int e = 0; e < 16; ++e) dst[e] = (__bf16)0.f;
  }
}

// ---------------------------------------------------------------------------
// Implicit-GEMM conv, bf16 WMMA, full-M blocking.
// Block = 128 threads = 4 waves; each wave owns 16 pixels x ALL cout blocks
// (MB accumulators). Per tap: cooperative uint4 copy of the tap's pre-packed
// A fragments into LDS, then per cin-block one B fragment (bf16 strided
// loads) feeds MB WMMAs.  mode 0: +bias(p0).  mode 1: BN(p0..p3)+SiLU.
// ---------------------------------------------------------------------------
template <int MB>
__global__ void conv_wmma_kernel(
    const __bf16* __restrict__ inb, long in_bstride, long in_coff, int cin,
    const __bf16* __restrict__ wpk,
    const float* __restrict__ p0, const float* __restrict__ p1,
    const float* __restrict__ p2, const float* __restrict__ p3,
    int mode, int k, int pad, int dil, int cout_real,
    float* __restrict__ out, long out_bstride,
    __bf16* __restrict__ outb, long outb_bstride)
{
  const int ntap = k * k;
  const int ncb  = cin >> 5;

  const int b     = blockIdx.x / 100;
  const int ptile = (blockIdx.x % 100) * 64;

  const int tid  = threadIdx.x;
  const int lane = tid & 31;
  const int wave = tid >> 5;

  extern __shared__ char smem[];
  __bf16* lds_a = (__bf16*)smem;                // ncb*MB frags * 512 bf16

  const __bf16* inp = inb + (long)b * in_bstride + in_coff;
  const int n  = lane & 15;
  const int p  = ptile + wave * 16 + n;
  const int y  = p / WW;
  const int x  = p % WW;
  const int kb = (lane < 16) ? 0 : 8;           // B-frag K half per lane

  v8f acc[MB] = {};

  const int fragsPerTap = ncb * MB;
  const int nvec = fragsPerTap * 64;            // uint4 (16B) chunks per tap

  for (int tap = 0; tap < ntap; ++tap) {
    // stage this tap's A fragments
    __syncthreads();
    {
      const uint4* src = (const uint4*)(wpk + (long)tap * fragsPerTap * 512);
      uint4* dst = (uint4*)lds_a;
      for (int i = tid; i < nvec; i += 128) dst[i] = src[i];
    }
    __syncthreads();

    const int iy = y - pad + (tap / k) * dil;
    const int ix = x - pad + (tap % k) * dil;
    const bool inr = (iy >= 0) & (iy < HH) & (ix >= 0) & (ix < WW);
    const long pix = (long)iy * WW + ix;

    // prefetch next tap's first input line (global_prefetch_b8)
    if (tap + 1 < ntap) {
      const int iy2 = y - pad + ((tap + 1) / k) * dil;
      const int ix2 = x - pad + ((tap + 1) % k) * dil;
      if ((iy2 >= 0) & (iy2 < HH) & (ix2 >= 0) & (ix2 < WW))
        __builtin_prefetch(inp + (long)kb * HW + (long)iy2 * WW + ix2, 0, 1);
    }

    for (int cb = 0; cb < ncb; ++cb) {
      v16bf bv;
      const __bf16* src2 = inp + (long)(cb * 32 + kb) * HW + pix;
      #pragma unroll
      for (int e = 0; e < 16; ++e)
        bv[e] = inr ? src2[(long)(e < 8 ? e : e + 8) * HW] : (__bf16)0.f;
      #pragma unroll
      for (int mb = 0; mb < MB; ++mb) {
        v16bf a = *(const v16bf*)(lds_a + (long)(cb * MB + mb) * 512 + lane * 16);
        acc[mb] = __builtin_amdgcn_wmma_f32_16x16x32_bf16(
            false, a, false, bv, (short)0, acc[mb], false, false);
      }
    }
  }

  // epilogue: D layout (ISA 7.12.2): VGPR r -> M = r + 8*(lane>=16)
  float* ob = out + (long)b * out_bstride;
  const int mbase = (lane < 16) ? 0 : 8;
  #pragma unroll
  for (int mb = 0; mb < MB; ++mb) {
    #pragma unroll
    for (int r = 0; r < 8; ++r) {
      const int cout = mb * 16 + mbase + r;
      if (cout < cout_real) {
        float v = acc[mb][r];
        if (mode == 1) {
          const float sc = p0[cout] * rsqrtf(p3[cout] + 1e-5f);
          v = (v - p2[cout]) * sc + p1[cout];
          v = v / (1.f + __expf(-v));            // SiLU
        } else {
          v += p0[cout];
        }
        ob[(long)cout * HW + p] = v;
        if (outb) outb[(long)b * outb_bstride + (long)cout * HW + p] = (__bf16)v;
      }
    }
  }
}

// ---------------------------------------------------------------------------
// Deformable bilinear sample + depthwise tap reduction (gather-bound).
// Corner weights (validity folded in) and clamped flat indices staged in LDS
// once per (tap, pixel); the 64-channel x 64-pixel loop is gathers + FMAs.
// Writes f32 result plus bf16 mirror for the downstream WMMA conv.
// ---------------------------------------------------------------------------
__global__ void sample_dw_kernel(
    const float* __restrict__ in, long in_bstride, long in_coff,
    const float* __restrict__ off,    // [b][2K][HW], dy=2*tap, dx=2*tap+1
    const float* __restrict__ dww,    // [64][K]
    float* __restrict__ out,          // [b][64][HW]
    __bf16* __restrict__ outb,        // bf16 mirror
    int k, int pad, int dil, int K)
{
  const int b     = blockIdx.x / 100;
  const int ptile = (blockIdx.x % 100) * 64;
  const int tid   = threadIdx.x;

  extern __shared__ char smem[];
  float* swt  = (float*)smem;                      // K*64*4 corner weights
  int*   sidx = (int*)(smem + (size_t)K * 64 * 4 * sizeof(float));

  const long offb = (long)b * (2L * K * HW);
  for (int it = tid; it < K * 64; it += 256) {
    const int tap = it / 64;
    const int pxl = it % 64;
    const int p   = ptile + pxl;
    const int y   = p / WW;
    const int x   = p % WW;
    const float dy = off[offb + (long)(2 * tap)     * HW + p];
    const float dx = off[offb + (long)(2 * tap + 1) * HW + p];
    const float py = (float)(y - pad + (tap / k) * dil) + dy;
    const float px = (float)(x - pad + (tap % k) * dil) + dx;
    const float fy = floorf(py), fx = floorf(px);
    const float ty = py - fy,    tx = px - fx;
    const int y0 = (int)fy, x0 = (int)fx;
    const float cw[4] = { (1.f - ty) * (1.f - tx), (1.f - ty) * tx,
                          ty * (1.f - tx),         ty * tx };
    const int base = it * 4;
    #pragma unroll
    for (int j = 0; j < 4; ++j) {
      const int cy = y0 + (j >> 1);
      const int cx = x0 + (j & 1);
      const bool v = (cy >= 0) & (cy < HH) & (cx >= 0) & (cx < WW);
      const int yc = min(max(cy, 0), HH - 1);
      const int xc = min(max(cx, 0), WW - 1);
      swt[base + j]  = v ? cw[j] : 0.f;
      sidx[base + j] = yc * WW + xc;
    }
  }
  __syncthreads();

  for (int wi = tid; wi < 64 * 64; wi += 256) {
    const int c   = wi >> 6;
    const int pxl = wi & 63;
    const float* xc  = in + (long)b * in_bstride + in_coff + (long)c * HW;
    const float* dwr = dww + (long)c * K;
    float acc = 0.f;
    for (int tap = 0; tap < K; ++tap) {
      const int base = (tap * 64 + pxl) * 4;
      const float s = swt[base]     * xc[sidx[base]]
                    + swt[base + 1] * xc[sidx[base + 1]]
                    + swt[base + 2] * xc[sidx[base + 2]]
                    + swt[base + 3] * xc[sidx[base + 3]];
      acc += dwr[tap] * s;
    }
    const long o = ((long)b * 64 + c) * HW + ptile + pxl;
    out[o]  = acc;
    outb[o] = (__bf16)acc;
  }
}

// ---------------------------------------------------------------------------
// Final residual fuse: out = x + u * concat(attnA, attnB)
// ---------------------------------------------------------------------------
__global__ void fuse_kernel(const float* __restrict__ x,
                            const float* __restrict__ u,
                            const float* __restrict__ aA,
                            const float* __restrict__ aB,
                            float* __restrict__ out, int total)
{
  const int i = blockIdx.x * 256 + threadIdx.x;
  if (i >= total) return;
  const int p = i % HW;
  const int c = (i / HW) & 127;
  const int b = i / (HW * 128);
  const float attn = (c < 64)
      ? aA[((long)b * 64 + c) * HW + p]
      : aB[((long)b * 64 + (c - 64)) * HW + p];
  out[i] = x[i] + u[i] * attn;
}

// ---------------------------------------------------------------------------
static void launch_packw(const float* w, __bf16* wpk, int cin, int k, int cr,
                         int MB, hipStream_t s)
{
  const int ntap = k * k, ncb = cin / 32;
  const int total = ntap * ncb * MB * 32;
  pack_w_kernel<<<dim3((total + 255) / 256), dim3(256), 0, s>>>(
      w, wpk, cin, ntap, cr, MB, ncb);
}

static void launch_conv(int MB, const __bf16* inb, long ibs, long icoff, int cin,
                        const __bf16* wpk, const float* q0, const float* q1,
                        const float* q2, const float* q3, int mode,
                        int k, int pad, int dil, int cr,
                        float* out, long obs, __bf16* outb, long obbs,
                        hipStream_t s)
{
  const size_t lds = (size_t)(cin / 32) * MB * 1024;
  dim3 g(NB * 100), blk(128);
  switch (MB) {
    case 2: conv_wmma_kernel<2><<<g, blk, lds, s>>>(inb, ibs, icoff, cin, wpk,
                q0, q1, q2, q3, mode, k, pad, dil, cr, out, obs, outb, obbs); break;
    case 4: conv_wmma_kernel<4><<<g, blk, lds, s>>>(inb, ibs, icoff, cin, wpk,
                q0, q1, q2, q3, mode, k, pad, dil, cr, out, obs, outb, obbs); break;
    case 7: conv_wmma_kernel<7><<<g, blk, lds, s>>>(inb, ibs, icoff, cin, wpk,
                q0, q1, q2, q3, mode, k, pad, dil, cr, out, obs, outb, obbs); break;
    default: conv_wmma_kernel<8><<<g, blk, lds, s>>>(inb, ibs, icoff, cin, wpk,
                q0, q1, q2, q3, mode, k, pad, dil, cr, out, obs, outb, obbs); break;
  }
}

static void launch_sample(const float* in, long ibs, long icoff,
                          const float* off, const float* dww,
                          float* out, __bf16* outb,
                          int k, int pad, int dil, hipStream_t s)
{
  const int K = k * k;
  const size_t lds = (size_t)K * 64 * 4 * (sizeof(float) + sizeof(int));
  sample_dw_kernel<<<dim3(NB * 100), dim3(256), lds, s>>>(
      in, ibs, icoff, off, dww, out, outb, k, pad, dil, K);
}

extern "C" void kernel_launch(void* const* d_in, const int* in_sizes, int n_in,
                              void* d_out, int out_size, void* d_ws,
                              size_t ws_size, hipStream_t stream)
{
  (void)in_sizes; (void)n_in; (void)out_size; (void)ws_size;
  const float* x      = (const float*)d_in[0];
  const float* cv1_w  = (const float*)d_in[1];
  const float* bn_g   = (const float*)d_in[2];
  const float* bn_b   = (const float*)d_in[3];
  const float* bn_m   = (const float*)d_in[4];
  const float* bn_v   = (const float*)d_in[5];
  const float* a0_ow  = (const float*)d_in[6];
  const float* a0_ob  = (const float*)d_in[7];
  const float* a0_w   = (const float*)d_in[8];
  const float* a1_ow  = (const float*)d_in[9];
  const float* a1_ob  = (const float*)d_in[10];
  const float* a1_w   = (const float*)d_in[11];
  const float* b0_ow  = (const float*)d_in[12];
  const float* b0_ob  = (const float*)d_in[13];
  const float* b0_w   = (const float*)d_in[14];
  const float* b1_ow  = (const float*)d_in[15];
  const float* b1_ob  = (const float*)d_in[16];
  const float* b1_w   = (const float*)d_in[17];
  const float* c1_w   = (const float*)d_in[18];
  const float* c1_b   = (const float*)d_in[19];
  const float* c2_w   = (const float*)d_in[20];
  const float* c2_b   = (const float*)d_in[21];

  const long S128 = 128L * HW, S64 = 64L * HW;
  const int  N128 = NB * 128 * HW;          // 3,276,800
  const int  N64  = NB * 64 * HW;           // 1,638,400

  // ---- workspace layout: f32 regions first, then 16B-aligned bf16 regions
  float* ws     = (float*)d_ws;
  float* ws_h   = ws;                        // N128
  float* ws_off = ws_h   + N128;             // up to 4*98*6400
  float* ws_t1  = ws_off + (long)NB * 98 * HW;
  float* ws_t2a = ws_t1  + N64;
  float* ws_t2b = ws_t2a + N64;
  float* ws_aA  = ws_t2b + N64;
  float* ws_aB  = ws_aA  + N64;

  __bf16* ws_xb   = (__bf16*)(ws_aB + N64);
  __bf16* ws_hb   = ws_xb   + N128;
  __bf16* ws_t1b  = ws_hb   + N128;
  __bf16* ws_t2ab = ws_t1b  + N64;
  __bf16* ws_t2bb = ws_t2ab + N64;
  __bf16* wp_cv1  = ws_t2bb + N64;           // 9*4*8*512
  __bf16* wp_a0   = wp_cv1  + 147456;        // 9*2*2*512
  __bf16* wp_a1   = wp_a0   + 18432;         // 25*2*4*512
  __bf16* wp_b0   = wp_a1   + 102400;
  __bf16* wp_b1   = wp_b0   + 102400;        // 49*2*7*512
  __bf16* wp_c1   = wp_b1   + 351232;        // 1*2*4*512
  __bf16* wp_c2   = wp_c1   + 4096;

  // ---- pack weights (fragment layout) and x (bf16 mirror)
  pack_bf16_kernel<<<dim3((N128 + 255) / 256), dim3(256), 0, stream>>>(x, ws_xb, N128);
  launch_packw(cv1_w, wp_cv1, 128, 3, 128, 8, stream);
  launch_packw(a0_ow, wp_a0,  64, 3, 18, 2, stream);
  launch_packw(a1_ow, wp_a1,  64, 5, 50, 4, stream);
  launch_packw(b0_ow, wp_b0,  64, 5, 50, 4, stream);
  launch_packw(b1_ow, wp_b1,  64, 7, 98, 7, stream);
  launch_packw(c1_w,  wp_c1,  64, 1, 64, 4, stream);
  launch_packw(c2_w,  wp_c2,  64, 1, 64, 4, stream);

  // 1) h = SiLU(BN(conv3x3(x)))  (f32 + bf16 mirror)
  launch_conv(8, ws_xb, S128, 0, 128, wp_cv1, bn_g, bn_b, bn_m, bn_v, 1,
              3, 1, 1, 128, ws_h, S128, ws_hb, S128, stream);

  // 2) branch a: deform k=3 then k=5/d=3, then 1x1
  launch_conv(2, ws_hb, S128, 0, 64, wp_a0, a0_ob, nullptr, nullptr, nullptr, 0,
              3, 1, 1, 18, ws_off, 18L * HW, nullptr, 0, stream);
  launch_sample(ws_h, S128, 0, ws_off, a0_w, ws_t1, ws_t1b, 3, 1, 1, stream);
  launch_conv(4, ws_t1b, S64, 0, 64, wp_a1, a1_ob, nullptr, nullptr, nullptr, 0,
              5, 6, 3, 50, ws_off, 50L * HW, nullptr, 0, stream);
  launch_sample(ws_t1, S64, 0, ws_off, a1_w, ws_t2a, ws_t2ab, 5, 6, 3, stream);
  launch_conv(4, ws_t2ab, S64, 0, 64, wp_c1, c1_b, nullptr, nullptr, nullptr, 0,
              1, 0, 1, 64, ws_aA, S64, nullptr, 0, stream);

  // 3) branch b: deform k=5 then k=7/d=3, then 1x1
  launch_conv(4, ws_hb, S128, S64, 64, wp_b0, b0_ob, nullptr, nullptr, nullptr, 0,
              5, 2, 1, 50, ws_off, 50L * HW, nullptr, 0, stream);
  launch_sample(ws_h, S128, S64, ws_off, b0_w, ws_t1, ws_t1b, 5, 2, 1, stream);
  launch_conv(7, ws_t1b, S64, 0, 64, wp_b1, b1_ob, nullptr, nullptr, nullptr, 0,
              7, 9, 3, 98, ws_off, 98L * HW, nullptr, 0, stream);
  launch_sample(ws_t1, S64, 0, ws_off, b1_w, ws_t2b, ws_t2bb, 7, 9, 3, stream);
  launch_conv(4, ws_t2bb, S64, 0, 64, wp_c2, c2_b, nullptr, nullptr, nullptr, 0,
              1, 0, 1, 64, ws_aB, S64, nullptr, 0, stream);

  // 4) out = x + u * concat(a, b)
  fuse_kernel<<<dim3((N128 + 255) / 256), dim3(256), 0, stream>>>(
      x, ws_h, ws_aA, ws_aB, (float*)d_out, N128);
}
